// MultiHeadAttention_33225867002596
// MI455X (gfx1250) — matrix operand order repaired and checked by reading, not compiled, not run on hardware
//
#include <hip/hip_runtime.h>
#include <hip/hip_bf16.h>

// ---------------- problem constants ----------------
#define B_   2
#define S_   2048
#define D_   2048
#define H_   16
#define DK_  128

typedef unsigned short u16;
typedef __attribute__((ext_vector_type(16))) __bf16 v16bf;
typedef __attribute__((ext_vector_type(8)))  float  v8f;
typedef unsigned int v4u_ __attribute__((ext_vector_type(4)));

// TDM builtin availability (this toolchain: clang-23 6-arg form)
#if defined(__has_builtin)
#  if __has_builtin(__builtin_amdgcn_tensor_load_to_lds) && \
      __has_builtin(__builtin_amdgcn_s_wait_tensorcnt)
#    define USE_TDM 1
#  endif
#endif
#ifndef USE_TDM
#  define USE_TDM 0
#endif

// ---------------- bf16 helpers ----------------
__device__ inline u16 f2bf(float f) {
  unsigned int x = __float_as_uint(f);
  unsigned int r = x + 0x7FFFu + ((x >> 16) & 1u);   // RNE
  return (u16)(r >> 16);
}
__device__ inline float bf2f(u16 u) {
  return __uint_as_float(((unsigned int)u) << 16);
}
__device__ inline v8f vzero8() {
  v8f z;
  #pragma unroll
  for (int i = 0; i < 8; ++i) z[i] = 0.0f;
  return z;
}

// ---------------- WMMA fragment (16 bf16 elems / lane) ----------------
struct Frag16 {
  union { v16bf v; u16 s[16]; unsigned int u[8]; };
};

// A fragment: 16x32 bf16 tile, row-major source, leading dim `ld`.
// Lanes 0-15 -> M=lane, K base 0; lanes 16-31 -> M=lane-16, K base 8.
// VGPR pairs 0-3: K = kbase+{0..7}; pairs 4-7: K = kbase+16+{0..7}.
__device__ inline void load_a_frag(Frag16& f, const u16* src, int ld,
                                   int row0, int k0, int lane) {
  const int m     = row0 + (lane & 15);
  const int kbase = k0 + ((lane >> 4) << 3);
  const u16* p = src + (size_t)m * ld + kbase;
  #pragma unroll
  for (int i = 0; i < 8; ++i) {
    const int k = ((i >= 4) ? 16 : 0) + (i & 3) * 2;
    f.s[2 * i]     = p[k];
    f.s[2 * i + 1] = p[k + 1];
  }
}

// B fragment: 32x16 (KxN) tile, K-major source. Lane l holds column n0+(l&15),
// K = 8*(l>>4)+{0..7} then 16+8*(l>>4)+{0..7} (same order as A).
__device__ inline void load_b_frag(Frag16& f, const u16* src, int ld,
                                   int k0, int n0, int lane) {
  const u16* p = src + (size_t)k0 * ld + n0 + (lane & 15);
  #pragma unroll
  for (int i = 0; i < 16; ++i) {
    const int k = ((i >= 8) ? 16 : 0) + ((lane >> 4) << 3) + (i & 7);
    f.s[i] = p[(size_t)k * ld];
  }
}

// Transposed B fragment via plain 128-bit LDS loads:
// logical B(k,n) = LDS[n][k] (16-bit elems, tile at (k0,n0), leading dim `ld`).
// Two ds_load_b128 (one per 16-wide K half) replace 16 scalar ds_load_u16.
// Lane l supplies the 16B segment at row n0+(l&15), col-half (l>>4)*8.
__device__ inline void load_bt_frag_tr(Frag16& f, unsigned lds_base_bytes,
                                       int ld, int k0, int n0, int lane) {
  const unsigned row  = (unsigned)(n0 + (lane & 15));
  const unsigned colh = (unsigned)((lane >> 4) * 8);
  const unsigned addr0 = lds_base_bytes + (row * (unsigned)ld + (unsigned)k0 + colh) * 2u;
  const unsigned addr1 = addr0 + 32u;   // K+16 half of the 32-deep fragment
  v4u_ r0, r1;
  asm volatile("ds_load_b128 %0, %1" : "=v"(r0) : "v"(addr0));
  asm volatile("ds_load_b128 %0, %1" : "=v"(r1) : "v"(addr1));
  asm volatile("s_wait_dscnt 0x0" ::: "memory");
  f.u[0] = r0[0]; f.u[1] = r0[1]; f.u[2] = r0[2]; f.u[3] = r0[3];
  f.u[4] = r1[0]; f.u[5] = r1[1]; f.u[6] = r1[2]; f.u[7] = r1[3];
}

__device__ inline v8f wmma_bf16(const Frag16& a, const Frag16& b, v8f c) {
  return __builtin_amdgcn_wmma_f32_16x16x32_bf16(
      false, a.v, false, b.v, (short)0, c, false, false);
}

// ---------------- Tensor Data Mover 2D tile load ----------------
#if USE_TDM
typedef unsigned int tdm_v4u __attribute__((ext_vector_type(4)));
typedef int          tdm_v8i __attribute__((ext_vector_type(8)));
typedef int          tdm_v4i __attribute__((ext_vector_type(4)));

// Load a rows x cols tile of 2-byte elements (row stride = stride elems)
// from global `gaddr` into LDS byte offset `lds_off`.
__device__ inline void tdm_load_tile_2d(unsigned lds_off, const void* gaddr,
                                        int rows, int cols, int stride) {
  const unsigned long long ga = (unsigned long long)(size_t)gaddr;
  tdm_v4u g0;
  g0[0] = 1u;                                            // count=1, user mode
  g0[1] = lds_off;                                       // lds_addr [63:32]
  g0[2] = (unsigned)(ga & 0xFFFFFFFFu);                  // global_addr lo
  g0[3] = (unsigned)((ga >> 32) & 0x1FFFFFFu) | (2u << 30); // ga hi | type=2
  tdm_v8i g1;
  g1[0] = 0x00010000;                                    // data_size=1 (2B)
  g1[1] = (int)((cols & 0xFFFF) << 16);                  // tensor_dim0 lo16
  g1[2] = (int)(((unsigned)cols >> 16) | ((rows & 0xFFFF) << 16)); // td0 hi | td1 lo
  g1[3] = (int)(((unsigned)rows >> 16) | ((cols & 0xFFFF) << 16)); // td1 hi | tile_dim0
  g1[4] = (int)(rows & 0xFFFF);                          // tile_dim1 | tile_dim2=0
  g1[5] = stride;                                        // tensor_dim0_stride lo32
  g1[6] = 0;
  g1[7] = 0;
  tdm_v4i z4 = {0, 0, 0, 0};
  tdm_v8i z8 = {0, 0, 0, 0, 0, 0, 0, 0};
  // clang-23 / therock-10.0 signature: (v4u, v8i, v4i, v4i, v8i, i32 cpol)
  __builtin_amdgcn_tensor_load_to_lds(g0, g1, z4, z4, z8, 0);
}
#endif

// ---------------- f32 -> bf16 conversion (8 elems / thread) ----------------
__global__ void cvt_f32_bf16_kernel(const float* __restrict__ src,
                                    u16* __restrict__ dst, size_t n8) {
  const size_t i = (size_t)blockIdx.x * blockDim.x + threadIdx.x;
  if (i >= n8) return;
  const size_t base = i * 8;
  const float4* s4 = (const float4*)(src + base);
  const float4 a = s4[0], b = s4[1];
  uint4 o;
  o.x = (unsigned)f2bf(a.x) | ((unsigned)f2bf(a.y) << 16);
  o.y = (unsigned)f2bf(a.z) | ((unsigned)f2bf(a.w) << 16);
  o.z = (unsigned)f2bf(b.x) | ((unsigned)f2bf(b.y) << 16);
  o.w = (unsigned)f2bf(b.z) | ((unsigned)f2bf(b.w) << 16);
  *(uint4*)(dst + base) = o;
}

// ---------------- bf16 GEMM: C = A(MxK) @ W(KxN) + bias ----------------
// Block tile 128x128, 256 threads = 8 waves (4x2), wave tile 32x64 (2x4 WMMA),
// K panel 64 (two 32-deep WMMA sub-steps per barrier pair).
#define BM 128
#define BN 128
#define BK 64

template<int WRITE_F32>
__global__ void __launch_bounds__(256)
gemm_bf16_kernel(const u16* __restrict__ A, const u16* __restrict__ W,
                 const float* __restrict__ bias,
                 u16* __restrict__ Cb, float* __restrict__ Cf,
                 int M, int N, int K) {
  __shared__ u16 As[BM * BK];
  __shared__ u16 Bs[BK * BN];

  const int tid  = threadIdx.x;
  const int lane = tid & 31;
  const int w    = tid >> 5;
  const int wm   = w >> 1;        // 0..3
  const int wn   = w & 1;         // 0..1
  const int m0   = blockIdx.y * BM;
  const int n0   = blockIdx.x * BN;

  v8f acc[2][4];
  #pragma unroll
  for (int i = 0; i < 2; ++i)
    #pragma unroll
    for (int j = 0; j < 4; ++j) acc[i][j] = vzero8();

  // staging coordinates: A 128x64 (2 threads/row, 32 elems each),
  //                      B 64x128 (4 threads/row, 32 elems each)
  const int ar = tid >> 1, ac = (tid & 1) * 32;
  const int br = tid >> 2, bc = (tid & 3) * 32;

  for (int kt = 0; kt < K; kt += BK) {
    {
      const u16* srcA = A + (size_t)(m0 + ar) * K + kt + ac;
      u16* dstA = As + ar * BK + ac;
      #pragma unroll
      for (int i = 0; i < 32; ++i) dstA[i] = srcA[i];
      const u16* srcB = W + (size_t)(kt + br) * N + n0 + bc;
      u16* dstB = Bs + br * BN + bc;
      #pragma unroll
      for (int i = 0; i < 32; ++i) dstB[i] = srcB[i];
      if (kt + BK < K) {  // prefetch next panels into L2 (global_prefetch_b8)
        __builtin_prefetch(A + (size_t)(m0 + ar) * K + kt + BK + ac, 0, 0);
        __builtin_prefetch(W + (size_t)(kt + BK + br) * N + n0 + bc, 0, 0);
      }
    }
    __syncthreads();

    #pragma unroll
    for (int ks = 0; ks < BK; ks += 32) {
      Frag16 a[2], b[4];
      #pragma unroll
      for (int i = 0; i < 2; ++i)
        load_a_frag(a[i], As, BK, wm * 32 + i * 16, ks, lane);
      #pragma unroll
      for (int j = 0; j < 4; ++j)
        load_b_frag(b[j], Bs, BN, ks, wn * 64 + j * 16, lane);
      #pragma unroll
      for (int i = 0; i < 2; ++i)
        #pragma unroll
        for (int j = 0; j < 4; ++j) acc[i][j] = wmma_bf16(a[i], b[j], acc[i][j]);
    }
    __syncthreads();
  }

  // epilogue: + bias, store (compile-time selected type)
  #pragma unroll
  for (int i = 0; i < 2; ++i) {
    #pragma unroll
    for (int j = 0; j < 4; ++j) {
      const int tm  = m0 + wm * 32 + i * 16;
      const int tn  = n0 + wn * 64 + j * 16;
      const int col = tn + (lane & 15);
      const float bv = bias[col];
      #pragma unroll
      for (int r = 0; r < 8; ++r) {
        const int row = tm + r + 8 * (lane >> 4);
        const float v = acc[i][j][r] + bv;
        if (WRITE_F32) Cf[(size_t)row * N + col] = v;
        else           Cb[(size_t)row * N + col] = f2bf(v);
      }
    }
  }
}

// ---------------- RoPE over (B,S,H,DK) bf16 tensor ----------------
__global__ void rope_kernel(u16* __restrict__ T) {
  const int idx = blockIdx.x * blockDim.x + threadIdx.x;  // B*S*H*64 = 2^22
  const int j = idx & 63;
  const int h = (idx >> 6) & (H_ - 1);
  const int s = (idx >> 10) & (S_ - 1);
  const int b = idx >> 21;
  const size_t base = ((size_t)(b * S_ + s)) * D_ + (size_t)h * DK_;
  const float inv = __expf(-(float)(2 * j) * (9.210340371976184f / 128.0f));
  const float ang = (float)s * inv;
  const float cs = __cosf(ang), sn = __sinf(ang);
  const float x1 = bf2f(T[base + j]);
  const float x2 = bf2f(T[base + j + 64]);
  T[base + j]      = f2bf(x1 * cs - x2 * sn);
  T[base + j + 64] = f2bf(x2 * cs + x1 * sn);
}

// ---------------- flash attention ----------------
// grid: (S/64, B*H), 256 threads = 8 waves.
// Dynamic LDS layout (byte offsets):
//   Qs @ 0      : 64x128 bf16  (16384 B)
//   Ks @ 16384  : 64x128 bf16  (16384 B)
//   Vs @ 32768  : 64x128 bf16  (16384 B)
//   Ss @ 49152  : 64x64  f32   (16384 B)
//   Ps @ 65536  : 64x64  bf16  ( 8192 B)
//   m/l/alpha   : 3*64 f32
#define LDS_OFF_K 16384u
#define LDS_OFF_V 32768u
#define ATTN_SMEM ((size_t)(3 * 64 * 128 * 2 + 64 * 64 * 4 + 64 * 64 * 2 + 3 * 64 * 4))

__global__ void __launch_bounds__(256)
attn_kernel(const u16* __restrict__ Q, const u16* __restrict__ Kt,
            const u16* __restrict__ V, u16* __restrict__ Ctx) {
  extern __shared__ char smem[];
  u16*   Qs   = (u16*)smem;              // 64*128
  u16*   Ks   = Qs + 64 * 128;           // 64*128
  u16*   Vs   = Ks + 64 * 128;           // 64*128
  float* Ss   = (float*)(Vs + 64 * 128); // 64*64 f32 scores
  u16*   Ps   = (u16*)(Ss + 64 * 64);    // 64*64 bf16 probs
  float* mrow = (float*)(Ps + 64 * 64);  // 64
  float* lrow = mrow + 64;               // 64
  float* arow = lrow + 64;               // 64

  const int tid  = threadIdx.x;
  const int lane = tid & 31;
  const int w    = tid >> 5;
  const int bh   = blockIdx.y;
  const int b    = bh >> 4;              // / H_
  const int h    = bh & (H_ - 1);
  const int q0   = blockIdx.x * 64;
  const size_t headoff = ((size_t)b * S_) * D_ + (size_t)h * DK_;
  const float scale = 0.08838834764831845f;   // 1/sqrt(128)

  { // load Q tile 64x128
    const int r = tid >> 2, c = (tid & 3) * 32;
    const u16* src = Q + headoff + (size_t)(q0 + r) * D_ + c;
    u16* dst = Qs + r * 128 + c;
    #pragma unroll
    for (int i = 0; i < 32; ++i) dst[i] = src[i];
  }
  if (tid < 64) { mrow[tid] = -1e30f; lrow[tid] = 0.0f; }
  __syncthreads();

  // O ownership: wave w -> row tile mt (16 rows), col tiles ng..ng+3 (64 cols)
  const int mt = w & 3;
  const int ng = (w >> 2) * 4;
  v8f o[4];
  #pragma unroll
  for (int j = 0; j < 4; ++j) o[j] = vzero8();

  for (int kb = 0; kb <= (int)blockIdx.x; ++kb) {
    const int k0g = kb * 64;
#if USE_TDM
    // DMA K/V 64x128 bf16 tiles into LDS via the Tensor Data Mover.
    if (tid == 0) {
      tdm_load_tile_2d(LDS_OFF_K, Kt + headoff + (size_t)k0g * D_, 64, 128, D_);
      tdm_load_tile_2d(LDS_OFF_V, V  + headoff + (size_t)k0g * D_, 64, 128, D_);
      __builtin_amdgcn_s_wait_tensorcnt(0);
    }
#else
    {
      const int r = tid >> 2, c = (tid & 3) * 32;
      const u16* sk = Kt + headoff + (size_t)(k0g + r) * D_ + c;
      const u16* sv = V  + headoff + (size_t)(k0g + r) * D_ + c;
      u16* dk = Ks + r * 128 + c;
      u16* dv = Vs + r * 128 + c;
      #pragma unroll
      for (int i = 0; i < 32; ++i) dk[i] = sk[i];
      #pragma unroll
      for (int i = 0; i < 32; ++i) dv[i] = sv[i];
    }
#endif
    __syncthreads();

    // S = scale * Q @ K^T  (64x64); wave computes 2 of 16 tiles
    {
      const int mt2 = w >> 1;
      #pragma unroll
      for (int jj = 0; jj < 2; ++jj) {
        const int nt2 = (w & 1) * 2 + jj;
        v8f sacc = vzero8();
        #pragma unroll
        for (int kc = 0; kc < 4; ++kc) {
          Frag16 a, bt;
          load_a_frag(a, Qs, 128, mt2 * 16, kc * 32, lane);
          load_bt_frag_tr(bt, LDS_OFF_K, 128, kc * 32, nt2 * 16, lane);
          sacc = wmma_bf16(a, bt, sacc);
        }
        const int coln = nt2 * 16 + (lane & 15);
        const int kpos = k0g + coln;
        #pragma unroll
        for (int r = 0; r < 8; ++r) {
          const int rowm = mt2 * 16 + r + 8 * (lane >> 4);
          const int qpos = q0 + rowm;
          float val = sacc[r] * scale;
          if (kpos > qpos) val = -1e30f;   // causal mask
          Ss[rowm * 64 + coln] = val;
        }
      }
    }
    __syncthreads();

    // online softmax row update: 4 threads per row (lane groups of 4, wave32)
    {
      const int row = tid >> 2;        // 0..63
      const int seg = tid & 3;         // 16 cols each
      const float mold = mrow[row];
      float mx = mold;
      #pragma unroll
      for (int c = 0; c < 16; ++c)
        mx = fmaxf(mx, Ss[row * 64 + seg * 16 + c]);
      mx = fmaxf(mx, __shfl_xor(mx, 1));
      mx = fmaxf(mx, __shfl_xor(mx, 2));
      float sum = 0.0f;
      #pragma unroll
      for (int c = 0; c < 16; ++c) {
        const float p = __expf(Ss[row * 64 + seg * 16 + c] - mx);
        sum += p;
        Ps[row * 64 + seg * 16 + c] = f2bf(p);
      }
      sum += __shfl_xor(sum, 1);
      sum += __shfl_xor(sum, 2);
      if (seg == 0) {
        const float alpha = __expf(mold - mx);
        mrow[row] = mx;
        lrow[row] = lrow[row] * alpha + sum;
        arow[row] = alpha;
      }
    }
    __syncthreads();

    // O = alpha*O + P @ V
    #pragma unroll
    for (int j = 0; j < 4; ++j) {
      #pragma unroll
      for (int r = 0; r < 8; ++r) {
        const int rowm = mt * 16 + r + 8 * (lane >> 4);
        o[j][r] *= arow[rowm];
      }
      #pragma unroll
      for (int kc = 0; kc < 2; ++kc) {
        Frag16 a, bb;
        load_a_frag(a, Ps, 64, mt * 16, kc * 32, lane);
        load_b_frag(bb, Vs, 128, kc * 32, (ng + j) * 16, lane);
        o[j] = wmma_bf16(a, bb, o[j]);
      }
    }
    __syncthreads();
  }

  // finalize: O /= l, store bf16 context (B*S, D) at head column offset
  #pragma unroll
  for (int j = 0; j < 4; ++j) {
    const int coln = (ng + j) * 16 + (lane & 15);
    #pragma unroll
    for (int r = 0; r < 8; ++r) {
      const int rowm = mt * 16 + r + 8 * (lane >> 4);
      const float v = o[j][r] / lrow[rowm];
      Ctx[headoff + (size_t)(q0 + rowm) * D_ + coln] = f2bf(v);
    }
  }
}

// ---------------- host launcher ----------------
extern "C" void kernel_launch(void* const* d_in, const int* in_sizes, int n_in,
                              void* d_out, int out_size, void* d_ws, size_t ws_size,
                              hipStream_t stream) {
  (void)in_sizes; (void)n_in; (void)out_size; (void)ws_size;

  const float* x  = (const float*)d_in[0];
  // d_in[1] = mask (causal, implicit in kernel)
  const float* Wq = (const float*)d_in[2];
  const float* bq = (const float*)d_in[3];
  const float* Wk = (const float*)d_in[4];
  const float* bk = (const float*)d_in[5];
  const float* Wv = (const float*)d_in[6];
  const float* bv = (const float*)d_in[7];
  const float* Wo = (const float*)d_in[8];
  const float* bo = (const float*)d_in[9];
  float* out = (float*)d_out;

  const size_t nx = (size_t)B_ * S_ * D_;   // 8,388,608
  const size_t nw = (size_t)D_ * D_;        // 4,194,304

  u16* p   = (u16*)d_ws;
  u16* xb  = p; p += nx;
  u16* wqb = p; p += nw;
  u16* wkb = p; p += nw;
  u16* wvb = p; p += nw;
  u16* wob = p; p += nw;
  u16* Qb  = p; p += nx;
  u16* Kb  = p; p += nx;
  u16* Vb  = p; p += nx;
  u16* Cx  = p; p += nx;

  const int thr = 256;
  cvt_f32_bf16_kernel<<<(unsigned)(nx / 8 / thr), thr, 0, stream>>>(x,  xb,  nx / 8);
  cvt_f32_bf16_kernel<<<(unsigned)(nw / 8 / thr), thr, 0, stream>>>(Wq, wqb, nw / 8);
  cvt_f32_bf16_kernel<<<(unsigned)(nw / 8 / thr), thr, 0, stream>>>(Wk, wkb, nw / 8);
  cvt_f32_bf16_kernel<<<(unsigned)(nw / 8 / thr), thr, 0, stream>>>(Wv, wvb, nw / 8);
  cvt_f32_bf16_kernel<<<(unsigned)(nw / 8 / thr), thr, 0, stream>>>(Wo, wob, nw / 8);

  dim3 gg(D_ / BN, (B_ * S_) / BM);   // 16 x 32
  gemm_bf16_kernel<0><<<gg, 256, 0, stream>>>(xb, wqb, bq, Qb, nullptr,
                                              B_ * S_, D_, D_);
  gemm_bf16_kernel<0><<<gg, 256, 0, stream>>>(xb, wkb, bk, Kb, nullptr,
                                              B_ * S_, D_, D_);
  gemm_bf16_kernel<0><<<gg, 256, 0, stream>>>(xb, wvb, bv, Vb, nullptr,
                                              B_ * S_, D_, D_);

  const int nrope = B_ * S_ * H_ * 64;      // 2^22
  rope_kernel<<<nrope / thr, thr, 0, stream>>>(Qb);
  rope_kernel<<<nrope / thr, thr, 0, stream>>>(Kb);

  attn_kernel<<<dim3(S_ / 64, B_ * H_), 256, ATTN_SMEM, stream>>>(Qb, Kb, Vb, Cx);

  gemm_bf16_kernel<1><<<gg, 256, 0, stream>>>(Cx, wob, bo, nullptr, out,
                                              B_ * S_, D_, D_);
}
